// MaxPool_80736795230628
// MI455X (gfx1250) — matrix operand order repaired and checked by reading, not compiled, not run on hardware
//
#include <hip/hip_runtime.h>

// ---------------------------------------------------------------------------
// MaxPool-by-similarity for MI455X (gfx1250, wave32, WMMA).
//
// Per (b,m) tile: D = Wf16(128x128) x Xt_f16(128 x 96) via
// v_wmma_f32_16x16x32_f16; sim combined with exact fp32 x; 32-way argmax;
// fp32 gather.  HBM-bound by design (~415 MB total traffic -> ~18 us floor
// at 23.3 TB/s; f16 WMMA keeps the 25.8 GFLOP GEMM under that floor).
//
// The 48 KB x-tile is streamed global->LDS with the CDNA5 async copy path
// (GLOBAL_LOAD_ASYNC_TO_LDS_B128, ASYNCcnt), bypassing VGPRs.
// ---------------------------------------------------------------------------

typedef __attribute__((ext_vector_type(16))) _Float16 v16h;
typedef __attribute__((ext_vector_type(8)))  _Float16 v8h;
typedef __attribute__((ext_vector_type(4)))  _Float16 v4h;
typedef __attribute__((ext_vector_type(8)))  float    v8f;
typedef __attribute__((ext_vector_type(4)))  float    v4f;

#define NBR 32      // neighbors
#define C   128     // channels
#define D3  3       // vector dim
// Padded LDS strides (avoid 256B-stride bank conflicts; keep 16B alignment)
#define XF_STRIDE  388   // floats per neighbor row of fp32 x (384 + 4)
#define SIM_STRIDE 132   // floats per neighbor row of sim   (128 + 4)
#define WH_STRIDE  136   // halfs per W row                  (128 + 8)
#define XT_STRIDE  136   // halfs per (d,N) row of Xt        (128 + 8)

#if defined(__gfx1250__) && \
    __has_builtin(__builtin_amdgcn_global_load_async_to_lds_b128) && \
    __has_builtin(__builtin_amdgcn_s_wait_asynccnt)
#define USE_ASYNC_LDS 1
// The builtin takes pointers to a 128-bit GCC-vector payload type:
//   param0: AS(1) (global, printed as __device__ in diagnostics)
//   param1: AS(3) (LDS)
typedef int v4i_t __attribute__((vector_size(16)));
typedef __attribute__((address_space(1))) v4i_t gmem_v4i;
typedef __attribute__((address_space(3))) v4i_t lds_v4i;
#else
#define USE_ASYNC_LDS 0
#endif

union Frag16 { v16h v; v8h h[2]; };

__global__ __launch_bounds__(256)
void maxpool_sim_kernel(const float* __restrict__ x,
                        const float* __restrict__ W,
                        float* __restrict__ out)
{
    __shared__ __attribute__((aligned(16))) float    xf[NBR * XF_STRIDE];      // 49664 B, exact fp32 tile
    __shared__ __attribute__((aligned(16))) float    sim[NBR * SIM_STRIDE];    // 16896 B
    __shared__ __attribute__((aligned(16))) _Float16 Wh[C * WH_STRIDE];        // 34816 B, f16 W (A operand)
    __shared__ __attribute__((aligned(16))) _Float16 Xt[3 * NBR * XT_STRIDE];  // 26112 B, f16 x^T (B operand)

    const int tid = threadIdx.x;
    const int bm  = blockIdx.x;                         // 0..8191
    const float* xs = x + (size_t)bm * (NBR * C * D3);  // 12288 floats

    // ---- Phase 1: tile loads ------------------------------------------------
    // x tile: 3072 x 16B (96 float4 per neighbor row, padded LDS row stride).
#if USE_ASYNC_LDS
    // CDNA5 async DMA path: global -> LDS directly, tracked by ASYNCcnt.
    #pragma unroll
    for (int it = 0; it < 12; ++it) {
        int f = tid + 256 * it;          // float4 index 0..3071
        int n = f / 96;
        int r = f - n * 96;
        __builtin_amdgcn_global_load_async_to_lds_b128(
            (gmem_v4i*)(void*)(xs + (size_t)f * 4),
            (lds_v4i*)(void*)(xf + n * XF_STRIDE + r * 4),
            0, 0);
    }
#else
    #pragma unroll
    for (int it = 0; it < 12; ++it) {
        int f = tid + 256 * it;          // float4 index 0..3071
        int n = f / 96;
        int r = f - n * 96;
        v4f v = *(const v4f*)(xs + (size_t)f * 4);
        *(v4f*)(xf + n * XF_STRIDE + r * 4) = v;
    }
#endif
    // W: 4096 float4 -> f16 rows (32 float4 per row); overlaps the async DMA.
    #pragma unroll
    for (int it = 0; it < 16; ++it) {
        int u   = tid + 256 * it;        // 0..4095
        int row = u >> 5;
        int c4  = u & 31;
        v4f w = *(const v4f*)(W + (size_t)u * 4);
        v4h h;
        h[0] = (_Float16)w[0]; h[1] = (_Float16)w[1];
        h[2] = (_Float16)w[2]; h[3] = (_Float16)w[3];
        *(v4h*)(Wh + row * WH_STRIDE + c4 * 4) = h;
    }
#if USE_ASYNC_LDS
    __builtin_amdgcn_s_wait_asynccnt(0);   // our async transfers landed in LDS
#endif
    __syncthreads();                       // ...and every other wave's too

    // ---- Phase 2: build f16 B operand Xt[d][N][j] = x[N][j][d] ---------------
    #pragma unroll
    for (int it = 0; it < 16; ++it) {
        int p = tid + 256 * it;          // (N,j) pair 0..4095
        int n = p >> 7;
        int j = p & 127;
        const float* s = xf + n * XF_STRIDE + j * 3;
        Xt[(0 * NBR + n) * XT_STRIDE + j] = (_Float16)s[0];
        Xt[(1 * NBR + n) * XT_STRIDE + j] = (_Float16)s[1];
        Xt[(2 * NBR + n) * XT_STRIDE + j] = (_Float16)s[2];
    }
    __syncthreads();

    // ---- Phase 3: WMMA GEMM + sim --------------------------------------------
    // Output blocks: 8 (mi: channel rows) x 2 (nh: neighbor halves).
    // Each wave handles 2 blocks; per block 3 accumulators (one per d), K=128.
    const int wave = tid >> 5;
    const int lane = tid & 31;
    const int hi   = (lane >> 4) & 1;
    const int l15  = lane & 15;

    #pragma unroll
    for (int pp = 0; pp < 2; ++pp) {
        const int pi = wave + 8 * pp;    // 0..15
        const int mi = pi >> 1;
        const int nh = pi & 1;
        const int arow = mi * 16 + l15;  // W row for this lane's A fragment
        const int coln = nh * 16 + l15;  // neighbor column for B fragment

        v8f acc0 = {}, acc1 = {}, acc2 = {};
        #pragma unroll
        for (int k = 0; k < 4; ++k) {
            // A (16x32 f16): lane=row; 16 contiguous K starting at 16*hi
            Frag16 a;
            const _Float16* ap = Wh + arow * WH_STRIDE + k * 32 + 16 * hi;
            a.h[0] = *(const v8h*)(ap);
            a.h[1] = *(const v8h*)(ap + 8);
            // B (32x16 f16): lane=column; 16 contiguous K starting at 16*hi
            Frag16 b0, b1, b2;
            const _Float16* bp0 = Xt + (0 * NBR + coln) * XT_STRIDE + k * 32 + 16 * hi;
            const _Float16* bp1 = Xt + (1 * NBR + coln) * XT_STRIDE + k * 32 + 16 * hi;
            const _Float16* bp2 = Xt + (2 * NBR + coln) * XT_STRIDE + k * 32 + 16 * hi;
            b0.h[0] = *(const v8h*)bp0; b0.h[1] = *(const v8h*)(bp0 + 8);
            b1.h[0] = *(const v8h*)bp1; b1.h[1] = *(const v8h*)(bp1 + 8);
            b2.h[0] = *(const v8h*)bp2; b2.h[1] = *(const v8h*)(bp2 + 8);

            acc0 = __builtin_amdgcn_wmma_f32_16x16x32_f16(false, a.v, false, b0.v,
                                                          (short)0, acc0, false, false);
            acc1 = __builtin_amdgcn_wmma_f32_16x16x32_f16(false, a.v, false, b1.v,
                                                          (short)0, acc1, false, false);
            acc2 = __builtin_amdgcn_wmma_f32_16x16x32_f16(false, a.v, false, b2.v,
                                                          (short)0, acc2, false, false);
        }
        // D layout: VGPR r <-> i_local = r + 8*hi, lane15 <-> column.
        // sim[N,i] = sum_d directions[i,N,d] * x[N,i,d]  (x read in exact fp32)
        const int nn = nh * 16 + l15;
        #pragma unroll
        for (int r = 0; r < 8; ++r) {
            const int i = mi * 16 + r + 8 * hi;
            const float* xr = xf + nn * XF_STRIDE + i * 3;
            sim[nn * SIM_STRIDE + i] =
                acc0[r] * xr[0] + acc1[r] * xr[1] + acc2[r] * xr[2];
        }
    }
    __syncthreads();

    // ---- Phase 4: 32-way argmax per channel + fp32 gather --------------------
    if (tid < C) {
        const int i = tid;
        float best = sim[i];
        int   bidx = 0;
        #pragma unroll
        for (int n = 1; n < NBR; ++n) {
            float s = sim[n * SIM_STRIDE + i];
            if (s > best) { best = s; bidx = n; }   // strict '>' => first max (jnp.argmax)
        }
        const float* src = xf + bidx * XF_STRIDE + i * 3;
        float* dst = out + ((size_t)bm * C + i) * 3;
        dst[0] = src[0]; dst[1] = src[1]; dst[2] = src[2];
    }
}

extern "C" void kernel_launch(void* const* d_in, const int* in_sizes, int n_in,
                              void* d_out, int out_size, void* d_ws, size_t ws_size,
                              hipStream_t stream) {
    const float* x = (const float*)d_in[0];
    const float* W = (const float*)d_in[1];
    float* out = (float*)d_out;
    const int tiles = in_sizes[0] / (NBR * C * D3);   // 2*4096 = 8192
    maxpool_sim_kernel<<<tiles, 256, 0, stream>>>(x, W, out);
}
